// LSTMAttentionDecoder_47622597378738
// MI455X (gfx1250) — compile-verified, matrix operand order checked by reading, not run on hardware
//
#include <hip/hip_runtime.h>
#include <hip/hip_bf16.h>

#define TY 128
#define BB 64
#define TXX 400
#define INN 512
#define HH 512
#define CC 512
#define NHH 8
#define DHH 64

typedef __attribute__((ext_vector_type(16))) _Float16 v16h;
typedef __attribute__((ext_vector_type(8)))  _Float16 v8h;
typedef __attribute__((ext_vector_type(8)))  float    v8f;

// ---------------------------------------------------------------------------
// f32 -> f16 convert
// ---------------------------------------------------------------------------
__global__ void f2h_kernel(const float* __restrict__ src, _Float16* __restrict__ dst, int n) {
    int i = blockIdx.x * 256 + threadIdx.x;
    if (i < n) dst[i] = (_Float16)src[i];
}

// ---------------------------------------------------------------------------
// init state: h, c, h16, coverage acc
// ---------------------------------------------------------------------------
__global__ void init_state_kernel(const float* __restrict__ h0, const float* __restrict__ c0,
                                  const float* __restrict__ cov0,
                                  float* __restrict__ h, float* __restrict__ c,
                                  _Float16* __restrict__ h16, float* __restrict__ acc) {
    int i = blockIdx.x * 256 + threadIdx.x;
    if (i < BB * HH) { h[i] = h0[i]; c[i] = c0[i]; h16[i] = (_Float16)h0[i]; }
    if (i < BB * TXX) acc[i] = cov0[i];
}

// ---------------------------------------------------------------------------
// WMMA fragment load for NT GEMM from row-major [rows, K] f16 array.
// 16-bit 16x32 A layout (ISA 7.12.2): lane r(+16*hi) holds row r,
// halves 0..7 = K (k0 + hi*8 .. +7), halves 8..15 = K (k0 + 16 + hi*8 .. +7).
// B fragment for NT (weight stored [N,K]) loads with the identical pattern.
// ---------------------------------------------------------------------------
__device__ __forceinline__ v16h load_frag_nt(const _Float16* __restrict__ base,
                                             int ld, int row0, int k0) {
    int lane = threadIdx.x & 31;
    int r  = lane & 15;
    int hi = lane >> 4;
    const _Float16* p = base + (size_t)(row0 + r) * ld + k0 + hi * 8;
    v8h lo = *reinterpret_cast<const v8h*>(p);
    v8h hv = *reinterpret_cast<const v8h*>(p + 16);
    v16h out;
#pragma unroll
    for (int i = 0; i < 8; ++i) { out[i] = lo[i]; out[i + 8] = hv[i]; }
    return out;
}

#define WMMA_F16(A_, B_, C_) \
    __builtin_amdgcn_wmma_f32_16x16x32_f16(false, (A_), false, (B_), (short)0, (C_), false, false)

// ---------------------------------------------------------------------------
// NT GEMM: C[m,n] = sum_k A[m,k] * W[n,k] (+ bias[n]) (+= old C if ACC)
// grid = (N/64, M/64), block = 128 (4 waves). Each wave: 16x64 strip.
// Software-pipelined: fragments for k+32 are fetched into fresh registers
// before the 4 WMMAs for k issue, so loads/WMMA overlap (partial loadcnt waits).
// ---------------------------------------------------------------------------
template<bool OUT_HALF, bool ACC>
__global__ __launch_bounds__(128)
void gemm_nt_wmma(const _Float16* __restrict__ A, const _Float16* __restrict__ W,
                  const float* __restrict__ bias,
                  float* __restrict__ Cf, _Float16* __restrict__ Ch,
                  int M, int N, int K) {
    int wave = threadIdx.x >> 5;
    int m0 = blockIdx.y * 64 + wave * 16;
    int n0 = blockIdx.x * 64;
    v8f acc[4] = {v8f{}, v8f{}, v8f{}, v8f{}};

    v16h a  = load_frag_nt(A, K, m0, 0);
    v16h b0 = load_frag_nt(W, K, n0, 0);
    v16h b1 = load_frag_nt(W, K, n0 + 16, 0);
    v16h b2 = load_frag_nt(W, K, n0 + 32, 0);
    v16h b3 = load_frag_nt(W, K, n0 + 48, 0);

    for (int k = 32; k < K; k += 32) {
        __builtin_prefetch(A + (size_t)m0 * K + k + 32, 0, 3);
        __builtin_prefetch(W + (size_t)n0 * K + k + 32, 0, 3);
        v16h an  = load_frag_nt(A, K, m0, k);
        v16h bn0 = load_frag_nt(W, K, n0, k);
        v16h bn1 = load_frag_nt(W, K, n0 + 16, k);
        v16h bn2 = load_frag_nt(W, K, n0 + 32, k);
        v16h bn3 = load_frag_nt(W, K, n0 + 48, k);
        acc[0] = WMMA_F16(a, b0, acc[0]);
        acc[1] = WMMA_F16(a, b1, acc[1]);
        acc[2] = WMMA_F16(a, b2, acc[2]);
        acc[3] = WMMA_F16(a, b3, acc[3]);
        a = an; b0 = bn0; b1 = bn1; b2 = bn2; b3 = bn3;
    }
    acc[0] = WMMA_F16(a, b0, acc[0]);
    acc[1] = WMMA_F16(a, b1, acc[1]);
    acc[2] = WMMA_F16(a, b2, acc[2]);
    acc[3] = WMMA_F16(a, b3, acc[3]);

    int lane = threadIdx.x & 31;
    int mo = (lane >> 4) * 8;   // M offset: VGPR j -> M=j (lanes 0-15), j+8 (lanes 16-31)
    int nn = lane & 15;
#pragma unroll
    for (int t = 0; t < 4; ++t) {
        int n = n0 + t * 16 + nn;
        float bv = bias ? bias[n] : 0.0f;
#pragma unroll
        for (int j = 0; j < 8; ++j) {
            int m = m0 + mo + j;
            size_t idx = (size_t)m * N + n;
            float v = acc[t][j] + bv;
            if (ACC) v += Cf[idx];
            if (OUT_HALF) Ch[idx] = (_Float16)v;
            else          Cf[idx] = v;
        }
    }
}

// ---------------------------------------------------------------------------
// LSTM cell elementwise. gates [B, 4H] order i,f,g,o. Writes h/c state,
// f16 h (for W_hh GEMM), s16 = concat(h1,c1) f16, and outputs hs/cs/ss.
// ---------------------------------------------------------------------------
__device__ __forceinline__ float sigmoidf_(float x) { return 1.0f / (1.0f + expf(-x)); }

__global__ void lstm_cell_kernel(const float* __restrict__ gates,
                                 const float* __restrict__ ymask_t,   // [B]
                                 float* __restrict__ h, float* __restrict__ c,
                                 _Float16* __restrict__ h16, _Float16* __restrict__ s16,
                                 float* __restrict__ hs_t, float* __restrict__ cs_t,
                                 float* __restrict__ ss_t) {
    int i = blockIdx.x * 256 + threadIdx.x;
    if (i >= BB * HH) return;
    int b = i >> 9;          // /512
    int j = i & 511;
    const float* g = gates + (size_t)b * 2048;
    float ig = g[j], fg = g[512 + j], gg = g[1024 + j], og = g[1536 + j];
    float cp = c[i], hp = h[i];
    float c1 = sigmoidf_(fg) * cp + sigmoidf_(ig) * tanhf(gg);
    float h1 = sigmoidf_(og) * tanhf(c1);
    float ym = ymask_t[b];
    h1 = ym * h1 + (1.0f - ym) * hp;
    c1 = ym * c1 + (1.0f - ym) * cp;
    h[i] = h1; c[i] = c1;
    h16[i] = (_Float16)h1;
    s16[(size_t)b * 1024 + j]       = (_Float16)h1;
    s16[(size_t)b * 1024 + 512 + j] = (_Float16)c1;
    hs_t[i] = h1; cs_t[i] = c1; ss_t[i] = h1;
}

// ---------------------------------------------------------------------------
// Attention scores: e[tx,b,h] = sum_d tanh(pctx + hq + acc*Wcov)[h*64+d]*xm * U[h,d]
// block: 256 thr = 8 waves; grid (TX/8, B). Each wave handles one tx.
// Lane takes c = lane*16 .. lane*16+15 (single head h = lane>>2).
// ---------------------------------------------------------------------------
__global__ __launch_bounds__(256)
void attn_scores_kernel(const _Float16* __restrict__ pctx, const float* __restrict__ hq,
                        const float* __restrict__ acc, const float* __restrict__ Wcov,
                        const float* __restrict__ Uatt, const float* __restrict__ xmask,
                        float* __restrict__ e) {
    __shared__ float sh_hq[512], sh_wc[512], sh_u[512];
    int b = blockIdx.y;
    for (int i = threadIdx.x; i < 512; i += 256) {
        sh_hq[i] = hq[(size_t)b * 512 + i];
        sh_wc[i] = Wcov[i];
        sh_u[i]  = Uatt[i];
    }
    __syncthreads();
    int wave = threadIdx.x >> 5, lane = threadIdx.x & 31;
    int tx = blockIdx.x * 8 + wave;
    float a  = acc[(size_t)b * TXX + tx];
    float xm = xmask[(size_t)tx * BB + b];
    int c0 = lane * 16;
    const _Float16* pc = pctx + ((size_t)tx * BB + b) * 512 + c0;
    v8h p0 = *reinterpret_cast<const v8h*>(pc);
    v8h p1 = *reinterpret_cast<const v8h*>(pc + 8);
    float part = 0.0f;
#pragma unroll
    for (int j = 0; j < 8; ++j) {
        float u = tanhf((float)p0[j] + sh_hq[c0 + j] + a * sh_wc[c0 + j]) * xm;
        part += u * sh_u[c0 + j];
    }
#pragma unroll
    for (int j = 0; j < 8; ++j) {
        float u = tanhf((float)p1[j] + sh_hq[c0 + 8 + j] + a * sh_wc[c0 + 8 + j]) * xm;
        part += u * sh_u[c0 + 8 + j];
    }
    // head h = lane>>2 spans lanes 4h..4h+3
    part += __shfl_xor(part, 1, 32);
    part += __shfl_xor(part, 2, 32);
    if ((lane & 3) == 0)
        e[((size_t)tx * BB + b) * NHH + (lane >> 2)] = part;
}

// ---------------------------------------------------------------------------
// Softmax over TX per (b,h): w = exp(e-max)*xmask / (sum + 1e-6)
// grid = B*NH, block = 256
// ---------------------------------------------------------------------------
__global__ __launch_bounds__(256)
void softmax_tx_kernel(const float* __restrict__ e, const float* __restrict__ xmask,
                       float* __restrict__ w) {
    __shared__ float red[256];
    int b = blockIdx.x >> 3, h = blockIdx.x & 7;
    int t0 = threadIdx.x, t1 = threadIdx.x + 256;
    float e0 = (t0 < TXX) ? e[((size_t)t0 * BB + b) * NHH + h] : -1e30f;
    float e1 = (t1 < TXX) ? e[((size_t)t1 * BB + b) * NHH + h] : -1e30f;
    red[threadIdx.x] = fmaxf(e0, e1);
    __syncthreads();
    for (int s = 128; s; s >>= 1) {
        if (threadIdx.x < s) red[threadIdx.x] = fmaxf(red[threadIdx.x], red[threadIdx.x + s]);
        __syncthreads();
    }
    float m = red[0];
    __syncthreads();
    float w0 = (t0 < TXX) ? expf(e0 - m) * xmask[(size_t)t0 * BB + b] : 0.0f;
    float w1 = (t1 < TXX) ? expf(e1 - m) * xmask[(size_t)t1 * BB + b] : 0.0f;
    red[threadIdx.x] = w0 + w1;
    __syncthreads();
    for (int s = 128; s; s >>= 1) {
        if (threadIdx.x < s) red[threadIdx.x] += red[threadIdx.x + s];
        __syncthreads();
    }
    float inv = 1.0f / (red[0] + 1e-6f);
    if (t0 < TXX) w[((size_t)t0 * BB + b) * NHH + h] = w0 * inv;
    if (t1 < TXX) w[((size_t)t1 * BB + b) * NHH + h] = w1 * inv;
}

// ---------------------------------------------------------------------------
// Coverage update + dist/Cs outputs. dist[b,tx] = w[tx,b,0]; Cs = old acc.
// grid = B, block = 256
// ---------------------------------------------------------------------------
__global__ void coverage_kernel(const float* __restrict__ w, float* __restrict__ acc,
                                float* __restrict__ dists_t, float* __restrict__ Cs_t) {
    int b = blockIdx.x;
    for (int tx = threadIdx.x; tx < TXX; tx += 256) {
        float d = w[((size_t)tx * BB + b) * NHH];
        float a = acc[(size_t)b * TXX + tx];
        Cs_t[(size_t)b * TXX + tx]    = a;
        dists_t[(size_t)b * TXX + tx] = d;
        acc[(size_t)b * TXX + tx]     = a + d;
    }
}

// ---------------------------------------------------------------------------
// att[b,c] = sum_tx w[tx,b,c>>6] * pv[tx,b,c]  (f16 out for next GEMM)
// grid = B, block = 512; w staged in LDS.
// ---------------------------------------------------------------------------
__global__ __launch_bounds__(512)
void attn_value_kernel(const _Float16* __restrict__ pv, const float* __restrict__ w,
                       _Float16* __restrict__ attv) {
    __shared__ float sw[TXX * NHH];
    int b = blockIdx.x;
    for (int i = threadIdx.x; i < TXX * NHH; i += 512) {
        int tx = i >> 3, h = i & 7;
        sw[i] = w[((size_t)tx * BB + b) * NHH + h];
    }
    __syncthreads();
    int j = threadIdx.x;
    int h = j >> 6;
    const _Float16* p = pv + (size_t)b * 512 + j;
    float s = 0.0f;
    for (int tx = 0; tx < TXX; ++tx)
        s += sw[tx * NHH + h] * (float)p[(size_t)tx * BB * 512];
    attv[(size_t)b * 512 + j] = (_Float16)s;
}

// ---------------------------------------------------------------------------
// LayerNorm over C=512 per batch row. grid = B, block = 256 (2 elems/thread)
// ---------------------------------------------------------------------------
__global__ __launch_bounds__(256)
void layernorm_kernel(const float* __restrict__ x, const float* __restrict__ g,
                      const float* __restrict__ be, float* __restrict__ out) {
    __shared__ float red[256];
    int b = blockIdx.x, tid = threadIdx.x;
    float v0 = x[(size_t)b * 512 + tid];
    float v1 = x[(size_t)b * 512 + 256 + tid];
    red[tid] = v0 + v1;
    __syncthreads();
    for (int s = 128; s; s >>= 1) {
        if (tid < s) red[tid] += red[tid + s];
        __syncthreads();
    }
    float mu = red[0] * (1.0f / 512.0f);
    __syncthreads();
    float d0 = v0 - mu, d1 = v1 - mu;
    red[tid] = d0 * d0 + d1 * d1;
    __syncthreads();
    for (int s = 128; s; s >>= 1) {
        if (tid < s) red[tid] += red[tid + s];
        __syncthreads();
    }
    float inv = rsqrtf(red[0] * (1.0f / 512.0f) + 1e-5f);
    out[(size_t)b * 512 + tid]       = d0 * inv * g[tid] + be[tid];
    out[(size_t)b * 512 + 256 + tid] = d1 * inv * g[tid + 256] + be[tid + 256];
}

// ---------------------------------------------------------------------------
// Host orchestration
// ---------------------------------------------------------------------------
static inline int cdiv(int a, int b) { return (a + b - 1) / b; }

extern "C" void kernel_launch(void* const* d_in, const int* in_sizes, int n_in,
                              void* d_out, int out_size, void* d_ws, size_t ws_size,
                              hipStream_t stream) {
    const float* y_emb    = (const float*)d_in[0];
    const float* context  = (const float*)d_in[1];
    const float* h0       = (const float*)d_in[2];
    const float* c0       = (const float*)d_in[3];
    const float* x_mask   = (const float*)d_in[4];
    const float* y_mask   = (const float*)d_in[5];
    const float* init_cov = (const float*)d_in[6];
    const float* W_ih     = (const float*)d_in[7];
    const float* W_hh     = (const float*)d_in[8];
    const float* b_ih     = (const float*)d_in[9];
    const float* b_hh     = (const float*)d_in[10];
    const float* Wc_att   = (const float*)d_in[11];
    const float* b_att    = (const float*)d_in[12];
    const float* Wv_att   = (const float*)d_in[13];
    const float* bv_att   = (const float*)d_in[14];
    const float* W_comb   = (const float*)d_in[15];
    const float* U_att    = (const float*)d_in[16];
    const float* W_cov    = (const float*)d_in[17];
    const float* W_concat = (const float*)d_in[18];
    const float* ln_g     = (const float*)d_in[19];
    const float* ln_b     = (const float*)d_in[20];

    // workspace bump allocator
    char* wp = (char*)d_ws;
    auto alloc = [&](size_t bytes) -> void* {
        void* r = (void*)wp;
        wp += (bytes + 255) & ~(size_t)255;
        return r;
    };
    const size_t MTX = (size_t)TXX * BB;            // 25600
    _Float16* ctx_h   = (_Float16*)alloc(MTX * CC * 2);
    _Float16* y_h     = (_Float16*)alloc((size_t)TY * BB * INN * 2);
    _Float16* Wih_h   = (_Float16*)alloc((size_t)4 * HH * INN * 2);
    _Float16* Whh_h   = (_Float16*)alloc((size_t)4 * HH * HH * 2);
    _Float16* Wc_h    = (_Float16*)alloc((size_t)CC * CC * 2);
    _Float16* Wv_h    = (_Float16*)alloc((size_t)CC * CC * 2);
    _Float16* Wcomb_h = (_Float16*)alloc((size_t)CC * 2 * HH * 2);
    _Float16* Wcat_h  = (_Float16*)alloc((size_t)CC * CC * 2);
    _Float16* pctx_h  = (_Float16*)alloc(MTX * CC * 2);
    _Float16* pv_h    = (_Float16*)alloc(MTX * CC * 2);
    float*    gates   = (float*)   alloc((size_t)BB * 4 * HH * 4);
    float*    hbuf    = (float*)   alloc((size_t)BB * HH * 4);
    float*    cbuf    = (float*)   alloc((size_t)BB * HH * 4);
    _Float16* h16     = (_Float16*)alloc((size_t)BB * HH * 2);
    _Float16* s16     = (_Float16*)alloc((size_t)BB * 2 * HH * 2);
    float*    hq      = (float*)   alloc((size_t)BB * CC * 4);
    float*    ebuf    = (float*)   alloc(MTX * NHH * 4);
    float*    wbuf    = (float*)   alloc(MTX * NHH * 4);
    float*    accbuf  = (float*)   alloc((size_t)BB * TXX * 4);
    _Float16* attv_h  = (_Float16*)alloc((size_t)BB * CC * 2);
    float*    attc    = (float*)   alloc((size_t)BB * CC * 4);

    // output layout: hs, cs, ss(=hs), atts, dists, Cs
    float* out   = (float*)d_out;
    const size_t SZ_H = (size_t)TY * BB * HH;        // 4194304
    const size_t SZ_T = (size_t)TY * BB * TXX;       // 3276800
    float* hs    = out;
    float* cs    = out + SZ_H;
    float* ss    = out + 2 * SZ_H;
    float* atts  = out + 3 * SZ_H;
    float* dists = out + 4 * SZ_H;
    float* Cs    = out + 4 * SZ_H + SZ_T;

    // ---- conversions to f16 ----
    auto conv = [&](const float* s, _Float16* d, int n) {
        f2h_kernel<<<cdiv(n, 256), 256, 0, stream>>>(s, d, n);
    };
    conv(context, ctx_h,  (int)(MTX * CC));
    conv(y_emb,   y_h,    TY * BB * INN);
    conv(W_ih,    Wih_h,  4 * HH * INN);
    conv(W_hh,    Whh_h,  4 * HH * HH);
    conv(Wc_att,  Wc_h,   CC * CC);
    conv(Wv_att,  Wv_h,   CC * CC);
    conv(W_comb,  Wcomb_h, CC * 2 * HH);
    conv(W_concat, Wcat_h, CC * CC);

    init_state_kernel<<<cdiv(BB * HH, 256), 256, 0, stream>>>(
        h0, c0, init_cov, hbuf, cbuf, h16, accbuf);

    // ---- precompute pctx, pv (f16 output, stays L2-resident) ----
    {
        dim3 grid(CC / 64, (int)(MTX / 64));
        gemm_nt_wmma<true, false><<<grid, 128, 0, stream>>>(
            ctx_h, Wc_h, b_att, nullptr, pctx_h, (int)MTX, CC, CC);
        gemm_nt_wmma<true, false><<<grid, 128, 0, stream>>>(
            ctx_h, Wv_h, bv_att, nullptr, pv_h, (int)MTX, CC, CC);
    }

    // ---- sequential decode ----
    for (int t = 0; t < TY; ++t) {
        const _Float16* x_t = y_h + (size_t)t * BB * INN;
        const float* ym_t = y_mask + (size_t)t * BB;
        float* hs_t = hs + (size_t)t * BB * HH;
        float* cs_t = cs + (size_t)t * BB * HH;
        float* ss_t = ss + (size_t)t * BB * HH;
        float* atts_t  = atts  + (size_t)t * BB * CC;
        float* dists_t = dists + (size_t)t * BB * TXX;
        float* Cs_t    = Cs    + (size_t)t * BB * TXX;

        // gates = x@W_ih.T + b_ih, then += h@W_hh.T + b_hh
        {
            dim3 grid(4 * HH / 64, BB / 64);
            gemm_nt_wmma<false, false><<<grid, 128, 0, stream>>>(
                x_t, Wih_h, b_ih, gates, nullptr, BB, 4 * HH, INN);
            gemm_nt_wmma<false, true><<<grid, 128, 0, stream>>>(
                h16, Whh_h, b_hh, gates, nullptr, BB, 4 * HH, HH);
        }
        lstm_cell_kernel<<<cdiv(BB * HH, 256), 256, 0, stream>>>(
            gates, ym_t, hbuf, cbuf, h16, s16, hs_t, cs_t, ss_t);

        // hq = s @ W_comb.T   (M=64, N=512, K=1024)
        {
            dim3 grid(CC / 64, BB / 64);
            gemm_nt_wmma<false, false><<<grid, 128, 0, stream>>>(
                s16, Wcomb_h, nullptr, hq, nullptr, BB, CC, 2 * HH);
        }

        attn_scores_kernel<<<dim3(TXX / 8, BB), 256, 0, stream>>>(
            pctx_h, hq, accbuf, W_cov, U_att, x_mask, ebuf);
        softmax_tx_kernel<<<BB * NHH, 256, 0, stream>>>(ebuf, x_mask, wbuf);
        coverage_kernel<<<BB, 256, 0, stream>>>(wbuf, accbuf, dists_t, Cs_t);
        attn_value_kernel<<<BB, 512, 0, stream>>>(pv_h, wbuf, attv_h);

        // attc = attv @ W_concat.T
        {
            dim3 grid(CC / 64, BB / 64);
            gemm_nt_wmma<false, false><<<grid, 128, 0, stream>>>(
                attv_h, Wcat_h, nullptr, attc, nullptr, BB, CC, CC);
        }
        layernorm_kernel<<<BB, 256, 0, stream>>>(attc, ln_g, ln_b, atts_t);
    }
}
